// AttentionPair_24635932410112
// MI455X (gfx1250) — matrix operand order repaired and checked by reading, not compiled, not run on hardware
//
#include <hip/hip_runtime.h>
#include <math.h>

// Problem dims (fixed by the reference)
#define BB     32
#define T_DIM  4096
#define DV     512     // DIM_VECT
#define DA     256     // DIM_ATTN
#define DM     512     // DIM_ATTN_BID (2*DIM_ATTN)
#define APITCH 520     // bf16 elements per LDS row (1040 B; /4 = 260 ≡ 4 mod 64 banks)

typedef float  v8f   __attribute__((ext_vector_type(8)));
typedef __bf16 v16bf __attribute__((ext_vector_type(16)));
typedef int    vsi4  __attribute__((vector_size(16)));   // matches builtin's param type

struct Frag32 { uint4 lo, hi; };   // 32 bytes -> one v16bf fragment
static_assert(sizeof(Frag32) == 32, "frag size");

// round-to-nearest-even float -> bf16 bits
__device__ __forceinline__ unsigned f2bf(float x) {
  unsigned u = __float_as_uint(x);
  return (u + 0x7FFFu + ((u >> 16) & 1u)) >> 16;
}
__device__ __forceinline__ float bf2f(unsigned h) { return __uint_as_float(h << 16); }

// ---------------------------------------------------------------------------
// K0: split W_mat (f32 [256,512]) into bf16 hi/lo tables (done once, tiny).
// ---------------------------------------------------------------------------
__global__ __launch_bounds__(256) void wsplit_kernel(const float* __restrict__ W_mat,
                                                     unsigned short* __restrict__ Whi,
                                                     unsigned short* __restrict__ Wlo) {
  const int i = blockIdx.x * 256 + threadIdx.x;   // grid 512 -> 131072 elements
  const float x = W_mat[i];
  const unsigned h = f2bf(x);
  const unsigned l = f2bf(x - bf2f(h));
  Whi[i] = (unsigned short)h;
  Wlo[i] = (unsigned short)l;
}

// ---------------------------------------------------------------------------
// K1: t1[b][a] = dot(vector[b,:], W_vec[a,:])
// ---------------------------------------------------------------------------
__global__ __launch_bounds__(256) void t1_kernel(const float* __restrict__ vector,
                                                 const float* __restrict__ W_vec,
                                                 float* __restrict__ t1) {
  __shared__ float v[DV];
  const int b = blockIdx.x;
  const int a = threadIdx.x;
  for (int i = threadIdx.x; i < DV; i += 256) v[i] = vector[b * DV + i];
  __syncthreads();
  float s = 0.0f;
  const float4* w = (const float4*)(W_vec + (size_t)a * DV);
  #pragma unroll 4
  for (int k = 0; k < DV / 4; ++k) {
    float4 q = w[k];
    s += q.x * v[4 * k + 0] + q.y * v[4 * k + 1] + q.z * v[4 * k + 2] + q.w * v[4 * k + 3];
  }
  t1[b * DA + a] = s;
}

// ---------------------------------------------------------------------------
// K2: fused logits GEMM with bf16 hi/lo split WMMA (16x16x32 bf16, f32 acc).
// Block = 256 threads = 8 waves; one (b, 16-row t-tile) per block.
//   D += Ahi*Bhi + Ahi*Blo + Alo*Bhi   (error ~ Alo*Blo ~ 2^-16 relative)
// Fragment layouts per ISA 7.12.2 (16-bit):
//   A 16x32: lane%16 = M; per lane two contiguous 8-K runs at
//            k0 + half*8 and k0 + 16 + half*8
//   B 32x16: lane%16 = N; per lane the same two 8-K runs as A
//   C/D:     c[g] = C[M = g + 8*half][N = lane%16]
// ---------------------------------------------------------------------------
__global__ __launch_bounds__(256) void logits_kernel(const float* __restrict__ matrix,
                                                     const unsigned short* __restrict__ Whi,
                                                     const unsigned short* __restrict__ Wlo,
                                                     const float* __restrict__ W_attn,
                                                     const float* __restrict__ t1,
                                                     const long long* __restrict__ lens,
                                                     float* __restrict__ logits) {
  __shared__ float          rawf[16 * 512];     // raw f32 tile (32 KB)
  __shared__ unsigned short Ahi[16 * APITCH];   // bf16 hi tile
  __shared__ unsigned short Alo[16 * APITCH];   // bf16 lo tile
  __shared__ float          part[8 * 16];

  const int b  = blockIdx.y;
  const int t0 = blockIdx.x * 16;
  const long long len = lens[b];
  if ((long long)t0 >= len) return;             // fully-masked tile: skip

  const int tid = threadIdx.x;
  const float* gsrc = matrix + ((size_t)b * T_DIM + t0) * DM;

  // ---- Stage raw f32 tile into LDS (async DMA path if toolchain has it) ----
#if __has_builtin(__builtin_amdgcn_global_load_async_to_lds_b128)
  for (int idx = tid; idx < 16 * 128; idx += 256) {
    __builtin_amdgcn_global_load_async_to_lds_b128(
        (__attribute__((address_space(1))) vsi4*)(gsrc + idx * 4),
        (__attribute__((address_space(3))) vsi4*)&rawf[idx * 4],
        /*offset=*/0, /*cpol=*/0);
  }
 #if __has_builtin(__builtin_amdgcn_s_wait_asynccnt)
  __builtin_amdgcn_s_wait_asynccnt(0);
 #endif
#else
  for (int idx = tid; idx < 16 * 128; idx += 256)
    *(float4*)&rawf[idx * 4] = *(const float4*)(gsrc + idx * 4);
#endif
  __syncthreads();

  // ---- Split f32 -> bf16 hi/lo in LDS ----
  for (int idx = tid; idx < 16 * 128; idx += 256) {
    const int r = idx >> 7;          // row 0..15
    const int c = idx & 127;         // float4 col
    float4 v = *(const float4*)&rawf[idx * 4];
    unsigned hx = f2bf(v.x), hy = f2bf(v.y), hz = f2bf(v.z), hw = f2bf(v.w);
    unsigned lx = f2bf(v.x - bf2f(hx));
    unsigned ly = f2bf(v.y - bf2f(hy));
    unsigned lz = f2bf(v.z - bf2f(hz));
    unsigned lw = f2bf(v.w - bf2f(hw));
    *(uint2*)&Ahi[r * APITCH + c * 4] = make_uint2(hx | (hy << 16), hz | (hw << 16));
    *(uint2*)&Alo[r * APITCH + c * 4] = make_uint2(lx | (ly << 16), lz | (lw << 16));
  }
  __syncthreads();

  const int wave = tid >> 5;               // 0..7
  const int lane = tid & 31;
  const int m    = lane & 15;              // M for A rows, N for B rows
  const int half = lane >> 4;              // 0 or 1

  float accM[8];
  #pragma unroll
  for (int g = 0; g < 8; ++g) accM[g] = 0.0f;

  #pragma unroll
  for (int j = 0; j < 2; ++j) {
    const int a0 = wave * 32 + j * 16;     // N-tile base in [0,256)
    const unsigned short* bhrow = Whi + (size_t)(a0 + m) * DM;
    const unsigned short* blrow = Wlo + (size_t)(a0 + m) * DM;

    v8f acc = {};
    #pragma unroll 2
    for (int k0 = 0; k0 < DM; k0 += 32) {
      // A fragments: two 8-K chunks each (hi and lo tiles)
      uint4 ah0 = *(const uint4*)&Ahi[m * APITCH + k0 + half * 8];
      uint4 ah1 = *(const uint4*)&Ahi[m * APITCH + k0 + 16 + half * 8];
      uint4 al0 = *(const uint4*)&Alo[m * APITCH + k0 + half * 8];
      uint4 al1 = *(const uint4*)&Alo[m * APITCH + k0 + 16 + half * 8];
      // B fragments: two 8-K chunks at k0 + half*8 and k0 + 16 + half*8 (same as A)
      uint4 bh0 = *(const uint4*)&bhrow[k0 + half * 8];
      uint4 bh1 = *(const uint4*)&bhrow[k0 + 16 + half * 8];
      uint4 bl0 = *(const uint4*)&blrow[k0 + half * 8];
      uint4 bl1 = *(const uint4*)&blrow[k0 + 16 + half * 8];

      Frag32 fa_h; fa_h.lo = ah0; fa_h.hi = ah1;
      Frag32 fa_l; fa_l.lo = al0; fa_l.hi = al1;
      Frag32 fb_h; fb_h.lo = bh0; fb_h.hi = bh1;
      Frag32 fb_l; fb_l.lo = bl0; fb_l.hi = bl1;
      v16bf Ah = __builtin_bit_cast(v16bf, fa_h);
      v16bf Al = __builtin_bit_cast(v16bf, fa_l);
      v16bf Bh = __builtin_bit_cast(v16bf, fb_h);
      v16bf Bl = __builtin_bit_cast(v16bf, fb_l);

      acc = __builtin_amdgcn_wmma_f32_16x16x32_bf16(false, Ah, false, Bh, (short)0, acc, false, false);
      acc = __builtin_amdgcn_wmma_f32_16x16x32_bf16(false, Ah, false, Bl, (short)0, acc, false, false);
      acc = __builtin_amdgcn_wmma_f32_16x16x32_bf16(false, Al, false, Bh, (short)0, acc, false, false);
    }

    const float t1v = t1[b * DA + a0 + m];
    const float wa  = W_attn[a0 + m];
    #pragma unroll
    for (int g = 0; g < 8; ++g) {
      float r = fmaxf(acc[g] + t1v, 0.0f) * wa;   // relu + attn weight
      r += __shfl_xor(r, 1);
      r += __shfl_xor(r, 2);
      r += __shfl_xor(r, 4);
      r += __shfl_xor(r, 8);
      accM[g] += r;
    }
  }

  if (m == 0) {
    #pragma unroll
    for (int g = 0; g < 8; ++g) part[wave * 16 + g + half * 8] = accM[g];
  }
  __syncthreads();

  if (tid < 16) {
    float s = 0.0f;
    #pragma unroll
    for (int w = 0; w < 8; ++w) s += part[w * 16 + tid];
    logits[(size_t)b * T_DIM + t0 + tid] = s;
  }
}

// ---------------------------------------------------------------------------
// K3: per-row masked softmax (per-row max == reference's global max after
// normalization; the shift cancels exactly).
// ---------------------------------------------------------------------------
__global__ __launch_bounds__(256) void softmax_kernel(const float* __restrict__ logits,
                                                      const long long* __restrict__ lens,
                                                      float* __restrict__ attn) {
  __shared__ float red[256];
  const int b = blockIdx.x;
  const int tid = threadIdx.x;
  const long long len = lens[b];

  float mx = -INFINITY;
  for (int t = tid; t < T_DIM; t += 256)
    if (t < len) mx = fmaxf(mx, logits[(size_t)b * T_DIM + t]);
  red[tid] = mx;
  __syncthreads();
  for (int s = 128; s > 0; s >>= 1) {
    if (tid < s) red[tid] = fmaxf(red[tid], red[tid + s]);
    __syncthreads();
  }
  mx = red[0];
  __syncthreads();

  float sum = 0.0f;
  for (int t = tid; t < T_DIM; t += 256) {
    float e = 0.0f;
    if (t < len) e = expf(logits[(size_t)b * T_DIM + t] - mx);
    attn[(size_t)b * T_DIM + t] = e;
    sum += e;
  }
  red[tid] = sum;
  __syncthreads();
  for (int s = 128; s > 0; s >>= 1) {
    if (tid < s) red[tid] += red[tid + s];
    __syncthreads();
  }
  const float inv = 1.0f / red[0];
  for (int t = tid; t < T_DIM; t += 256)
    attn[(size_t)b * T_DIM + t] *= inv;
}

// ---------------------------------------------------------------------------
// K4: partial representations, grid (8 t-chunks, 32 b), deterministic partials.
// ---------------------------------------------------------------------------
__global__ __launch_bounds__(512) void reps_partial_kernel(const float* __restrict__ matrix,
                                                           const float* __restrict__ attn,
                                                           const long long* __restrict__ lens,
                                                           float* __restrict__ partial) {
  const int b = blockIdx.y;
  const int c = blockIdx.x;
  const int d = threadIdx.x;
  const long long len = lens[b];
  const int tstart = c * (T_DIM / 8);
  int tend = tstart + (T_DIM / 8);
  if ((long long)tend > len) tend = (int)((len > (long long)tstart) ? len : (long long)tstart);

  float acc = 0.0f;
  const float* mrow = matrix + ((size_t)b * T_DIM + tstart) * DM + d;
  for (int t = tstart; t < tend; ++t, mrow += DM) {
    __builtin_prefetch(mrow + 8 * DM, 0, 0);   // -> global_prefetch_b8
    acc += attn[(size_t)b * T_DIM + t] * (*mrow);
  }
  partial[((size_t)b * 8 + c) * DM + d] = acc;
}

__global__ __launch_bounds__(512) void reps_reduce_kernel(const float* __restrict__ partial,
                                                          float* __restrict__ reps) {
  const int b = blockIdx.x;
  const int d = threadIdx.x;
  float s = 0.0f;
  #pragma unroll
  for (int c = 0; c < 8; ++c) s += partial[((size_t)b * 8 + c) * DM + d];
  reps[(size_t)b * DM + d] = s;
}

// ---------------------------------------------------------------------------
extern "C" void kernel_launch(void* const* d_in, const int* in_sizes, int n_in,
                              void* d_out, int out_size, void* d_ws, size_t ws_size,
                              hipStream_t stream) {
  const float*     vector  = (const float*)d_in[0];      // [32,512]
  const float*     matrix  = (const float*)d_in[1];      // [32,4096,512]
  const long long* lens    = (const long long*)d_in[2];  // [32] int64
  const float*     W_vec   = (const float*)d_in[3];      // [256,512]
  const float*     W_mat   = (const float*)d_in[4];      // [256,512]
  const float*     W_attn  = (const float*)d_in[5];      // [1,256]

  float* out_reps = (float*)d_out;                       // [32,512]
  float* out_attn = (float*)d_out + BB * DM;             // [32,4096]

  // Workspace (bytes): t1 32KB | logits 512KB | partial 512KB | Whi 256KB | Wlo 256KB
  char* ws = (char*)d_ws;
  float*          t1      = (float*)ws;
  float*          logits  = (float*)(ws + 32768);
  float*          partial = (float*)(ws + 32768 + 524288);
  unsigned short* Whi     = (unsigned short*)(ws + 32768 + 524288 + 524288);
  unsigned short* Wlo     = (unsigned short*)(ws + 32768 + 524288 + 524288 + 262144);

  wsplit_kernel<<<(DA * DM) / 256, 256, 0, stream>>>(W_mat, Whi, Wlo);
  t1_kernel<<<BB, 256, 0, stream>>>(vector, W_vec, t1);
  logits_kernel<<<dim3(T_DIM / 16, BB), 256, 0, stream>>>(matrix, Whi, Wlo, W_attn, t1, lens, logits);
  softmax_kernel<<<BB, 256, 0, stream>>>(logits, lens, out_attn);
  reps_partial_kernel<<<dim3(8, BB), 512, 0, stream>>>(matrix, out_attn, lens, partial);
  reps_reduce_kernel<<<BB, 512, 0, stream>>>(partial, out_reps);
}
